// TimeAwareMHAttn_15015205667375
// MI455X (gfx1250) — compile-verified, matrix operand order checked by reading, not run on hardware
//
#include <hip/hip_runtime.h>

// ---------------------------------------------------------------------------
// Time-aware MHA for MI455X (gfx1250, wave32, WMMA, async global->LDS).
//
// Fusions:
//   scores = [Q|Tk] @ [K|Q]^T / 32 + mask      (inner dim 128 per head)
//   context = softmax(scores) @ (V + Tv)       (U = V+Tv fused at projection)
// All GEMMs run on v_wmma_f32_16x16x32_bf16 with fp32 accumulation.
// Dense GEMMs use GLOBAL_LOAD_ASYNC_TO_LDS_B128 double-buffering (ASYNCcnt);
// attention is flash-style (online softmax), never materializing S x S.
// ---------------------------------------------------------------------------

#define SQ     1024          // sequence length S
#define DMODEL 1024          // model dim D
#define NHEAD  16
#define DHEAD  64
#define BATCH  4
#define MROWS  (BATCH * SQ)  // 4096 rows for all dense GEMMs

typedef __bf16 bf16;
typedef __attribute__((ext_vector_type(16))) __bf16 v16bf;
typedef __attribute__((ext_vector_type(8)))  __bf16 v8bf;
typedef __attribute__((ext_vector_type(8)))  float  v8f;

__device__ inline bf16 f2bf(float f) { return (bf16)f; }

__device__ inline v16bf join8(v8bf lo, v8bf hi) {
  v16bf r;
#pragma unroll
  for (int i = 0; i < 8; ++i) { r[i] = lo[i]; r[i + 8] = hi[i]; }
  return r;
}

// Load a 16x32 bf16 fragment (A-layout; identical striping serves as the
// B-fragment when `mat` is the K-major transpose of B).  Per ISA 7.12.2:
// lane L holds row (L&15); elements 0..7 are K = 8*(L>>4)+e, elements
// 8..15 are K = 16 + 8*(L>>4) + (e-8).  Both halves are contiguous 16B.
__device__ inline v16bf frag_rm(const bf16* mat, int row0, int ld, int kb, int lane) {
  int m = lane & 15, h = lane >> 4;
  const bf16* p = mat + (size_t)(row0 + m) * ld + kb + 8 * h;
  v8bf lo = *(const v8bf*)p;
  v8bf hi = *(const v8bf*)(p + 16);
  return join8(lo, hi);
}

__device__ inline v8f wmma_bf16(v16bf a, v16bf b, v8f c) {
  // (neg_a, A, neg_b, B, c_mod, C, reuse_a, reuse_b)
  return __builtin_amdgcn_wmma_f32_16x16x32_bf16(false, a, false, b,
                                                 (short)0, c, false, false);
}

// Async DMA of one 16B chunk straight into LDS (no VGPR staging, ASYNCcnt).
__device__ inline void async_lds_b128(void* lds_ptr, const void* gptr) {
  unsigned lo = (unsigned)(size_t)lds_ptr;  // generic->LDS: low 32 bits
  asm volatile("global_load_async_to_lds_b128 %0, %1, off"
               :: "v"(lo), "v"(gptr) : "memory");
}
__device__ inline void wait_asynccnt_le4() {
  asm volatile("s_wait_asynccnt 0x4" ::: "memory");
}
__device__ inline void wait_asynccnt_0() {
  asm volatile("s_wait_asynccnt 0x0" ::: "memory");
}

// Butterfly reductions across the 16-lane half owning one score row.
__device__ inline float bfly_max16(float v) {
#pragma unroll
  for (int m = 1; m < 16; m <<= 1) v = fmaxf(v, __shfl_xor(v, m, 32));
  return v;
}
__device__ inline float bfly_sum16(float v) {
#pragma unroll
  for (int m = 1; m < 16; m <<= 1) v += __shfl_xor(v, m, 32);
  return v;
}

// ---------------------------------------------------------------------------
// Elementwise converts (run once, off the critical WMMA path)
// ---------------------------------------------------------------------------
__global__ void cvt_f32_to_bf16(const float* __restrict__ in,
                                bf16* __restrict__ out, int n) {
  int i = blockIdx.x * 256 + threadIdx.x;
  if (i < n) out[i] = f2bf(in[i]);
}

// W [K][N] fp32 -> Wt [N][K] bf16 so B-fragments load contiguously along K.
__global__ void transpose_cvt_w(const float* __restrict__ W,
                                bf16* __restrict__ Wt) {
  int idx = blockIdx.x * 256 + threadIdx.x;  // DMODEL*DMODEL threads
  int n = idx >> 10, k = idx & (DMODEL - 1);
  Wt[(size_t)n * DMODEL + k] = f2bf(W[(size_t)k * DMODEL + n]);
}

// ---------------------------------------------------------------------------
// Generic WMMA GEMM: C[m,n] = sum_k A1[m,k] W1[k,n] (+ A2 W2) + b1[n] (+b2[n])
// Block tile 128x128 (8 waves, each 2x4 WMMA tiles), K staged 32-wide in LDS
// with double-buffered async global->LDS DMA.  ~64 FLOP/B of global traffic.
// Epilogue scatters into attention-friendly per-head layouts.
// ---------------------------------------------------------------------------
enum { ST_HEAD128 = 0, ST_UT = 1, ST_F32 = 2 };

__global__ __launch_bounds__(256)
void gemm_bf16_wmma(const bf16* __restrict__ A1, const bf16* __restrict__ W1t,
                    const bf16* __restrict__ A2, const bf16* __restrict__ W2t,
                    const float* __restrict__ b1, const float* __restrict__ b2,
                    void* dst1, int mode1, int col01,
                    void* dst2, int mode2, int col02, int K) {
  __shared__ bf16 Als[2][128 * 32];   // double-buffered A tile
  __shared__ bf16 Bls[2][128 * 32];   // double-buffered Wt tile (N-major)
  int tid = threadIdx.x;
  int lane = tid & 31, wave = tid >> 5;
  int wm = wave & 3, wn = wave >> 2;          // 4x2 wave grid -> 32x64 / wave
  int m0 = blockIdx.x * 128;
  int n0 = blockIdx.y * 128;
  const int ktiles = K / 32;

  v8f c[2][4] = {};

  // One 128x32 + 128x32 LDS tile = 4 x 16B async chunks per thread.
  auto issue = [&](const bf16* A, const bf16* Wt, int k0, int buf) {
#pragma unroll
    for (int i = 0; i < 2; ++i) {
      int cc = tid + i * 256;
      int row = cc >> 2, off = (cc & 3) * 8;
      async_lds_b128(&Als[buf][row * 32 + off],
                     A + (size_t)(m0 + row) * K + k0 + off);
      async_lds_b128(&Bls[buf][row * 32 + off],
                     Wt + (size_t)(n0 + row) * K + k0 + off);
    }
  };
  auto compute = [&](int buf) {
    v16bf a0 = frag_rm(&Als[buf][0], wm * 32,      32, 0, lane);
    v16bf a1 = frag_rm(&Als[buf][0], wm * 32 + 16, 32, 0, lane);
#pragma unroll
    for (int jt = 0; jt < 4; ++jt) {
      v16bf g = frag_rm(&Bls[buf][0], wn * 64 + jt * 16, 32, 0, lane);
      c[0][jt] = wmma_bf16(a0, g, c[0][jt]);
      c[1][jt] = wmma_bf16(a1, g, c[1][jt]);
    }
  };
  // Branch-free steady state: always prefetch t+1 while computing t.
  auto run_pair = [&](const bf16* A, const bf16* Wt) {
    issue(A, Wt, 0, 0);
#pragma unroll 1
    for (int t = 0; t < ktiles - 1; ++t) {
      issue(A, Wt, (t + 1) * 32, (t + 1) & 1);
      wait_asynccnt_le4();          // in-order: 4 oldest (tile t) landed
      __syncthreads();              // tile t visible to all waves
      compute(t & 1);
      __syncthreads();              // all waves done before buf re-filled
    }
    wait_asynccnt_0();
    __syncthreads();
    compute((ktiles - 1) & 1);
    __syncthreads();
  };

  run_pair(A1, W1t);
  if (A2 != nullptr) run_pair(A2, W2t);   // uniform, once

  // Epilogue: resolve the (uniform) layout mode once, then unrolled stores.
  int mlane = lane & 15, h = lane >> 4;
  auto epilogue = [&](void* dst, int mode, int col0) {
    if (mode == ST_HEAD128) {       // [B,H][S][128] bf16, cols col0..col0+63
      bf16* d = (bf16*)dst;
#pragma unroll
      for (int it = 0; it < 2; ++it)
#pragma unroll
        for (int jt = 0; jt < 4; ++jt) {
          int ng = n0 + wn * 64 + jt * 16 + mlane;
          float bias = b1[ng] + (b2 ? b2[ng] : 0.0f);
          int hh = ng >> 6, dk = ng & (DHEAD - 1);
#pragma unroll
          for (int r = 0; r < 8; ++r) {
            int mg = m0 + wm * 32 + it * 16 + r + 8 * h;
            int b = mg >> 10, s = mg & (SQ - 1);
            d[((size_t)((b * NHEAD + hh) * SQ + s)) * 128 + col0 + dk] =
                f2bf(c[it][jt][r] + bias);
          }
        }
    } else if (mode == ST_UT) {     // U transposed: [B,H][DK][S] bf16
      bf16* d = (bf16*)dst;
#pragma unroll
      for (int it = 0; it < 2; ++it)
#pragma unroll
        for (int jt = 0; jt < 4; ++jt) {
          int ng = n0 + wn * 64 + jt * 16 + mlane;
          float bias = b1[ng] + (b2 ? b2[ng] : 0.0f);
          int hh = ng >> 6, dk = ng & (DHEAD - 1);
#pragma unroll
          for (int r = 0; r < 8; ++r) {
            int mg = m0 + wm * 32 + it * 16 + r + 8 * h;
            int b = mg >> 10, s = mg & (SQ - 1);
            d[((size_t)((b * NHEAD + hh) * DHEAD + dk)) * SQ + s] =
                f2bf(c[it][jt][r] + bias);
          }
        }
    } else {                        // plain fp32 [M][D]
      float* d = (float*)dst;
#pragma unroll
      for (int it = 0; it < 2; ++it)
#pragma unroll
        for (int jt = 0; jt < 4; ++jt) {
          int ng = n0 + wn * 64 + jt * 16 + mlane;
          float bias = b1[ng] + (b2 ? b2[ng] : 0.0f);
#pragma unroll
          for (int r = 0; r < 8; ++r) {
            int mg = m0 + wm * 32 + it * 16 + r + 8 * h;
            d[(size_t)mg * DMODEL + ng] = c[it][jt][r] + bias;
          }
        }
    }
  };
  epilogue(dst1, mode1, col01);
  if (dst2 != nullptr) epilogue(dst2, mode2, col02);
}

// ---------------------------------------------------------------------------
// Flash attention: one wave owns 16 query rows; loops keys 32 at a time.
//   Ahat = [Q|Tk]  (S x 128 per head),  Bhat = [K|Q]  (S x 128 per head)
//   Ut   = (V+Tv)^T (DK x S per head)   -> B-fragments contiguous along K=j
// Online softmax in registers; P transposed via private 1KB LDS tile
// (single-wave DS ops are in-order -> no barrier needed).
// ---------------------------------------------------------------------------
__global__ __launch_bounds__(128)
void attn_flash_wmma(const bf16* __restrict__ Ahat, const bf16* __restrict__ Bhat,
                     const bf16* __restrict__ Ut, const float* __restrict__ mask,
                     bf16* __restrict__ ctx) {
  __shared__ bf16 Pls[4][16 * 32];
  int lane = threadIdx.x & 31, wave = threadIdx.x >> 5;
  int bh = blockIdx.y;
  int b = bh >> 4, head = bh & 15;
  int i0 = blockIdx.x * 64 + wave * 16;
  const bf16* Abh = Ahat + (size_t)bh * SQ * 128;
  const bf16* Bbh = Bhat + (size_t)bh * SQ * 128;
  const bf16* Ubh = Ut   + (size_t)bh * DHEAD * SQ;
  int mlane = lane & 15, h = lane >> 4;

  v16bf aq[4];
#pragma unroll
  for (int cc = 0; cc < 4; ++cc) aq[cc] = frag_rm(Abh, i0, 128, 32 * cc, lane);

  v8f acc[4] = {};
  float mrow[8], lrow[8];
#pragma unroll
  for (int r = 0; r < 8; ++r) { mrow[r] = -1e30f; lrow[r] = 0.0f; }

  // Per-lane mask base: row (i0+8h), column mlane; mask is [B,1,S,S].
  const float* mbase = mask + ((size_t)b * SQ + (size_t)(i0 + 8 * h)) * SQ + mlane;

  for (int j0 = 0; j0 < SQ; j0 += 32) {
    if (j0 + 32 < SQ)  // pull next mask tile toward L2 (global_prefetch_b8)
      __builtin_prefetch(mbase + j0 + 32, 0, 0);
    v8f s[2];
#pragma unroll
    for (int jt = 0; jt < 2; ++jt) {
      v8f z = {};
#pragma unroll
      for (int cc = 0; cc < 4; ++cc)
        z = wmma_bf16(aq[cc], frag_rm(Bbh, j0 + 16 * jt, 128, 32 * cc, lane), z);
#pragma unroll
      for (int r = 0; r < 8; ++r)   // scale = 1/sqrt(D) = 1/32, then +mask
        z[r] = z[r] * 0.03125f + mbase[(size_t)r * SQ + j0 + 16 * jt];
      s[jt] = z;
    }
    // online softmax statistics per row (rows r+8h live in this 16-lane half)
    float alpha[8];
#pragma unroll
    for (int r = 0; r < 8; ++r) {
      float t = bfly_max16(fmaxf(s[0][r], s[1][r]));
      float mn = fmaxf(mrow[r], t);
      alpha[r] = __expf(mrow[r] - mn);
      mrow[r] = mn;
    }
#pragma unroll
    for (int jt = 0; jt < 2; ++jt)
#pragma unroll
      for (int r = 0; r < 8; ++r) s[jt][r] = __expf(s[jt][r] - mrow[r]);
#pragma unroll
    for (int r = 0; r < 8; ++r) {
      float rs = bfly_sum16(s[0][r] + s[1][r]);
      lrow[r] = lrow[r] * alpha[r] + rs;
    }
    // transpose P (C-layout -> A-layout) through this wave's LDS slice
#pragma unroll
    for (int jt = 0; jt < 2; ++jt)
#pragma unroll
      for (int r = 0; r < 8; ++r)
        Pls[wave][(r + 8 * h) * 32 + jt * 16 + mlane] = f2bf(s[jt][r]);
    v16bf ap = frag_rm(Pls[wave], 0, 32, 0, lane);
#pragma unroll
    for (int f = 0; f < 4; ++f) {
#pragma unroll
      for (int r = 0; r < 8; ++r) acc[f][r] *= alpha[r];
      acc[f] = wmma_bf16(ap, frag_rm(Ubh, 16 * f, SQ, j0, lane), acc[f]);
    }
  }
  // normalize and write context as bf16 [B*S][D] (heads recombined)
#pragma unroll
  for (int f = 0; f < 4; ++f)
#pragma unroll
    for (int r = 0; r < 8; ++r) {
      int gi = i0 + r + 8 * h;
      float v = acc[f][r] / lrow[r];
      ctx[((size_t)(b * SQ + gi)) * DMODEL + head * DHEAD + f * 16 + mlane] = f2bf(v);
    }
}

// ---------------------------------------------------------------------------
extern "C" void kernel_launch(void* const* d_in, const int* in_sizes, int n_in,
                              void* d_out, int out_size, void* d_ws, size_t ws_size,
                              hipStream_t stream) {
  (void)in_sizes; (void)n_in; (void)out_size; (void)ws_size;
  const float* q_in  = (const float*)d_in[0];
  const float* k_in  = (const float*)d_in[1];
  const float* v_in  = (const float*)d_in[2];
  const float* tk_in = (const float*)d_in[3];
  const float* tv_in = (const float*)d_in[4];
  const float* mask  = (const float*)d_in[5];
  const float* Wq  = (const float*)d_in[6];  const float* bq  = (const float*)d_in[7];
  const float* Wk  = (const float*)d_in[8];  const float* bk  = (const float*)d_in[9];
  const float* Wv  = (const float*)d_in[10]; const float* bv  = (const float*)d_in[11];
  const float* Wtk = (const float*)d_in[12]; const float* btk = (const float*)d_in[13];
  const float* Wtv = (const float*)d_in[14]; const float* btv = (const float*)d_in[15];
  const float* Wm  = (const float*)d_in[16]; const float* bm  = (const float*)d_in[17];

  char* ws = (char*)d_ws;
  size_t off = 0;
  auto alloc = [&](size_t bytes) -> void* {
    void* p = ws + off;
    off += (bytes + 255) & ~(size_t)255;
    return p;
  };
  const size_t ACT  = (size_t)MROWS * DMODEL * sizeof(bf16);    // 8 MB
  const size_t WMAT = (size_t)DMODEL * DMODEL * sizeof(bf16);   // 2 MB
  bf16* qx   = (bf16*)alloc(ACT);
  bf16* kx   = (bf16*)alloc(ACT);
  bf16* vx   = (bf16*)alloc(ACT);
  bf16* tkx  = (bf16*)alloc(ACT);
  bf16* tvx  = (bf16*)alloc(ACT);
  bf16* WqT  = (bf16*)alloc(WMAT);
  bf16* WkT  = (bf16*)alloc(WMAT);
  bf16* WvT  = (bf16*)alloc(WMAT);
  bf16* WtkT = (bf16*)alloc(WMAT);
  bf16* WtvT = (bf16*)alloc(WMAT);
  bf16* WmT  = (bf16*)alloc(WMAT);
  bf16* AhatB = (bf16*)alloc((size_t)BATCH * NHEAD * SQ * 128 * sizeof(bf16)); // 16 MB
  bf16* BhatB = (bf16*)alloc((size_t)BATCH * NHEAD * SQ * 128 * sizeof(bf16)); // 16 MB
  bf16* UtB   = (bf16*)alloc((size_t)BATCH * NHEAD * DHEAD * SQ * sizeof(bf16)); // 8 MB
  bf16* ctx   = (bf16*)alloc(ACT);                                               // 8 MB

  // 1) activations -> bf16
  {
    const int n = MROWS * DMODEL;
    const int g = (n + 255) / 256;
    cvt_f32_to_bf16<<<g, 256, 0, stream>>>(q_in,  qx,  n);
    cvt_f32_to_bf16<<<g, 256, 0, stream>>>(k_in,  kx,  n);
    cvt_f32_to_bf16<<<g, 256, 0, stream>>>(v_in,  vx,  n);
    cvt_f32_to_bf16<<<g, 256, 0, stream>>>(tk_in, tkx, n);
    cvt_f32_to_bf16<<<g, 256, 0, stream>>>(tv_in, tvx, n);
  }
  // 2) weights -> bf16 transposed [N][K]
  {
    const int g = (DMODEL * DMODEL) / 256;
    transpose_cvt_w<<<g, 256, 0, stream>>>(Wq,  WqT);
    transpose_cvt_w<<<g, 256, 0, stream>>>(Wk,  WkT);
    transpose_cvt_w<<<g, 256, 0, stream>>>(Wv,  WvT);
    transpose_cvt_w<<<g, 256, 0, stream>>>(Wtk, WtkT);
    transpose_cvt_w<<<g, 256, 0, stream>>>(Wtv, WtvT);
    transpose_cvt_w<<<g, 256, 0, stream>>>(Wm,  WmT);
  }
  // 3) projections (WMMA GEMMs) with fused per-head layouts
  dim3 ggrid(MROWS / 128, DMODEL / 128);
  // Q -> Ahat cols 0..63  AND  Bhat cols 64..127
  gemm_bf16_wmma<<<ggrid, 256, 0, stream>>>(qx, WqT, nullptr, nullptr, bq, nullptr,
                                            AhatB, ST_HEAD128, 0,
                                            BhatB, ST_HEAD128, 64, DMODEL);
  // Tk -> Ahat cols 64..127
  gemm_bf16_wmma<<<ggrid, 256, 0, stream>>>(tkx, WtkT, nullptr, nullptr, btk, nullptr,
                                            AhatB, ST_HEAD128, 64,
                                            nullptr, 0, 0, DMODEL);
  // K -> Bhat cols 0..63
  gemm_bf16_wmma<<<ggrid, 256, 0, stream>>>(kx, WkT, nullptr, nullptr, bk, nullptr,
                                            BhatB, ST_HEAD128, 0,
                                            nullptr, 0, 0, DMODEL);
  // U = value@Wv + time_v@Wtv + bv + btv  -> Ut (transposed per head)
  gemm_bf16_wmma<<<ggrid, 256, 0, stream>>>(vx, WvT, tvx, WtvT, bv, btv,
                                            UtB, ST_UT, 0,
                                            nullptr, 0, 0, DMODEL);
  // 4) flash attention
  attn_flash_wmma<<<dim3(SQ / 64, BATCH * NHEAD), 128, 0, stream>>>(
      AhatB, BhatB, UtB, mask, ctx);
  // 5) output projection -> fp32 d_out
  gemm_bf16_wmma<<<ggrid, 256, 0, stream>>>(ctx, WmT, nullptr, nullptr, bm, nullptr,
                                            d_out, ST_F32, 0,
                                            nullptr, 0, 0, DMODEL);
}
